// GCN_27324581937408
// MI455X (gfx1250) — matrix operand order repaired and checked by reading, not compiled, not run on hardware
//
#include <hip/hip_runtime.h>
#include <cmath>

#define NNODES 100000
#define NFEAT  512
#define NHID   128
#define NCLASS 40
#define NLAYER 4
#define ALPHA_ 0.1f
#define LAMDA_ 0.5f

typedef __attribute__((ext_vector_type(16))) __bf16        v16bf;
typedef __attribute__((ext_vector_type(8)))  float         v8f;
typedef __attribute__((ext_vector_type(4)))  unsigned int  v4u;
typedef __attribute__((ext_vector_type(8)))  int           v8i;
typedef __attribute__((ext_vector_type(4)))  int           v4i;

// ---------------------------------------------------------------------------
// Pack an f32 weight matrix W[K][Ncols] (row-major) into per-lane bf16
// B-fragments for v_wmma_f32_16x16x32_bf16.
// Linear layout: ((s*ntN + t)*32 + L)*16 + e   where
//   k = s*32 + (L>>4)*8 + (e&7) + (e>>3)*16 ,  n = t*16 + (L&15)
// ---------------------------------------------------------------------------
__global__ void pack_w_kernel(const float* __restrict__ W, __bf16* __restrict__ out,
                              int K, int Ncols, int ntN) {
    int idx = blockIdx.x * blockDim.x + threadIdx.x;
    int total = (K / 32) * ntN * 32 * 16;
    if (idx >= total) return;
    int e = idx & 15;
    int L = (idx >> 4) & 31;
    int tp = idx >> 9;
    int t = tp % ntN;
    int s = tp / ntN;
    int k = s * 32 + ((L >> 4) * 8) + (e & 7) + ((e >> 3) * 16);
    int n = t * 16 + (L & 15);
    float v = (n < Ncols) ? W[(long)k * Ncols + n] : 0.0f;
    out[idx] = (__bf16)v;
}

// ---------------------------------------------------------------------------
// Stage `ndwords` dwords from global `src` into LDS at `smem` using the
// Tensor Data Mover (one DMA per workgroup), falling back to a cooperative
// copy if the TDM builtin is unavailable. Must be called by ALL threads.
// ---------------------------------------------------------------------------
__device__ __forceinline__ void stage_to_lds(const void* src, void* smem, int ndwords)
{
#if __has_builtin(__builtin_amdgcn_tensor_load_to_lds) && __has_builtin(__builtin_amdgcn_s_wait_tensorcnt)
    if ((threadIdx.x >> 5) == 0) {               // one wave issues the DMA
        unsigned long long ga = (unsigned long long)(uintptr_t)src;
        unsigned int lds_off  = (unsigned int)(uintptr_t)smem;   // addr[31:0] = LDS offset
        unsigned int nd = (unsigned int)ndwords;

        v4u g0;
        g0[0] = (unsigned)__builtin_amdgcn_readfirstlane(1);                 // count=1, user
        g0[1] = (unsigned)__builtin_amdgcn_readfirstlane((int)lds_off);      // lds_addr
        g0[2] = (unsigned)__builtin_amdgcn_readfirstlane((int)(ga & 0xffffffffu));
        g0[3] = (unsigned)__builtin_amdgcn_readfirstlane(
                    (int)(((ga >> 32) & 0x01ffffffu) | (2u << 30)));         // addr[56:32] | type=2

        v8i g1;
        g1[0] = __builtin_amdgcn_readfirstlane((int)(2u << 16));             // data_size=4B
        g1[1] = __builtin_amdgcn_readfirstlane((int)((nd & 0xffffu) << 16)); // tensor_dim0[15:0]
        g1[2] = __builtin_amdgcn_readfirstlane((int)(((nd >> 16) & 0xffffu) | (1u << 16))); // dim0 hi | tensor_dim1=1
        g1[3] = __builtin_amdgcn_readfirstlane((int)((nd & 0xffffu) << 16)); // tile_dim0 (fits 16b)
        g1[4] = __builtin_amdgcn_readfirstlane(1);                           // tile_dim1=1, tile_dim2=0
        g1[5] = __builtin_amdgcn_readfirstlane((int)nd);                     // tensor_dim0_stride[31:0]
        g1[6] = __builtin_amdgcn_readfirstlane((int)((nd & 0xffffu) << 16)); // stride0 hi=0 | stride1[15:0]
        g1[7] = __builtin_amdgcn_readfirstlane((int)(nd >> 16));             // stride1[47:16]

        v4i g2; g2[0] = 0; g2[1] = 0; g2[2] = 0; g2[3] = 0;
        v4i g3 = g2;
        v8i g4; g4[0] = 0; g4[1] = 0; g4[2] = 0; g4[3] = 0;
                g4[4] = 0; g4[5] = 0; g4[6] = 0; g4[7] = 0;
        __builtin_amdgcn_tensor_load_to_lds(g0, g1, g2, g3, g4, 0);
        __builtin_amdgcn_s_wait_tensorcnt(0);
    }
    __syncthreads();
#else
    const uint4* s4 = (const uint4*)src;
    uint4* d4 = (uint4*)smem;
    int n4 = ndwords >> 2;
    for (int i = threadIdx.x; i < n4; i += blockDim.x) d4[i] = s4[i];
    __syncthreads();
#endif
}

// ---------------------------------------------------------------------------
// bf16 WMMA GEMM: each wave computes a 16-row strip x (NT*16) cols.
// A is f32 in global memory (converted to bf16 in-register); B is pre-packed
// and staged into LDS by the Tensor Data Mover once per workgroup.
// MODE 0: out = relu(acc + bias[n])                      (fc1)
// MODE 1: out = relu(theta*acc + (1-theta)*support[m,n]) (conv layers)
// MODE 2: out[m*ncolsOut+n] = acc + bias[n], n<ncolsOut  (fc2 -> logits)
// ---------------------------------------------------------------------------
template <int NT, int MODE, int KSTEPS>
__global__ __launch_bounds__(256) void gemm_wmma_kernel(
    const float* __restrict__ A, int lda,
    const __bf16* __restrict__ Bp,
    float* __restrict__ out, int ldo,
    const float* __restrict__ bias,
    const float* __restrict__ support,
    float theta, int M, int ncolsOut)
{
    extern __shared__ __bf16 smem[];                     // KSTEPS*NT*1024 bytes
    stage_to_lds(Bp, smem, KSTEPS * NT * 128);           // dwords = KSTEPS*NT*32*16/4, half of packed B

    int wave = blockIdx.x * (blockDim.x >> 5) + (threadIdx.x >> 5);
    int m0 = wave * 16;
    if (m0 >= M) return;                                 // wave-uniform
    int L = threadIdx.x & 31;
    int mrow = m0 + (L & 15);
    int koff = (L >> 4) * 8;

    v8f acc[NT] = {};
    const v16bf* __restrict__ BsV = (const v16bf*)smem;

    #pragma unroll
    for (int s = 0; s < KSTEPS; ++s) {
        const float* ap = A + (long)mrow * lda + s * 32 + koff;
        if (s + 1 < KSTEPS) __builtin_prefetch(ap + 32, 0, 3);   // global_prefetch_b8
        float4 a0 = *(const float4*)(ap + 0);
        float4 a1 = *(const float4*)(ap + 4);
        float4 a2 = *(const float4*)(ap + 16);
        float4 a3 = *(const float4*)(ap + 20);
        v16bf af;
        af[0]  = (__bf16)a0.x; af[1]  = (__bf16)a0.y; af[2]  = (__bf16)a0.z; af[3]  = (__bf16)a0.w;
        af[4]  = (__bf16)a1.x; af[5]  = (__bf16)a1.y; af[6]  = (__bf16)a1.z; af[7]  = (__bf16)a1.w;
        af[8]  = (__bf16)a2.x; af[9]  = (__bf16)a2.y; af[10] = (__bf16)a2.z; af[11] = (__bf16)a2.w;
        af[12] = (__bf16)a3.x; af[13] = (__bf16)a3.y; af[14] = (__bf16)a3.z; af[15] = (__bf16)a3.w;
        #pragma unroll
        for (int t = 0; t < NT; ++t) {
            v16bf bf = BsV[(s * NT + t) * 32 + L];       // ds_load_b128 x2
            acc[t] = __builtin_amdgcn_wmma_f32_16x16x32_bf16(
                false, af, false, bf, (short)0, acc[t], false, false);
        }
    }

    #pragma unroll
    for (int t = 0; t < NT; ++t) {
        int n = t * 16 + (L & 15);
        #pragma unroll
        for (int v = 0; v < 8; ++v) {
            int m = m0 + v + 8 * (L >> 4);
            float val = acc[t][v];
            if (MODE == 0) {
                val += bias[n];
                val = fmaxf(val, 0.0f);
                out[(long)m * ldo + n] = val;
            } else if (MODE == 1) {
                float sup = support[(long)m * ldo + n];
                val = theta * val + (1.0f - theta) * sup;
                val = fmaxf(val, 0.0f);
                out[(long)m * ldo + n] = val;
            } else {
                if (n < ncolsOut) {
                    val += bias[n];
                    out[(long)m * ncolsOut + n] = val;
                }
            }
        }
    }
}

// ---------------------------------------------------------------------------
// SpMM: hi += vals * h[cols] scattered to rows. Rows are sorted, so each wave
// walks a contiguous edge span, accumulates 4 feats/lane in registers while
// the row is unchanged, and flushes with f32 atomics on row change.
// ---------------------------------------------------------------------------
#define SPMM_SPAN 128

__global__ __launch_bounds__(256) void spmm_kernel(
    const int* __restrict__ rows, const int* __restrict__ cols,
    const float* __restrict__ vals, const float* __restrict__ h,
    float* __restrict__ hi, int nE)
{
    int wave = blockIdx.x * (blockDim.x >> 5) + (threadIdx.x >> 5);
    int lane = threadIdx.x & 31;
    long e0 = (long)wave * SPMM_SPAN;
    if (e0 >= nE) return;
    long e1 = e0 + SPMM_SPAN;
    if (e1 > nE) e1 = nE;

    float ax = 0.f, ay = 0.f, az = 0.f, aw = 0.f;
    int curRow = -1;
    for (long e = e0; e < e1; ++e) {
        int r = rows[e];
        if (r != curRow) {
            if (curRow >= 0) {
                float* dst = hi + (long)curRow * NHID + lane * 4;
                atomicAdd(dst + 0, ax); atomicAdd(dst + 1, ay);
                atomicAdd(dst + 2, az); atomicAdd(dst + 3, aw);
            }
            curRow = r;
            ax = ay = az = aw = 0.f;
        }
        int c = cols[e];
        float w = vals[e];
        float4 hv = *(const float4*)(h + (long)c * NHID + lane * 4);
        ax += w * hv.x; ay += w * hv.y; az += w * hv.z; aw += w * hv.w;
    }
    if (curRow >= 0) {
        float* dst = hi + (long)curRow * NHID + lane * 4;
        atomicAdd(dst + 0, ax); atomicAdd(dst + 1, ay);
        atomicAdd(dst + 2, az); atomicAdd(dst + 3, aw);
    }
}

__global__ void zero_kernel(float* __restrict__ p, long n) {
    long i = (long)blockIdx.x * blockDim.x + threadIdx.x;
    if (i < n) p[i] = 0.0f;
}

// support = (1-alpha)*hi + alpha*h0, in place on hi
__global__ void support_kernel(float* __restrict__ hi, const float* __restrict__ h0, long n) {
    long i = (long)blockIdx.x * blockDim.x + threadIdx.x;
    if (i < n) hi[i] = (1.0f - ALPHA_) * hi[i] + ALPHA_ * h0[i];
}

// one wave per node; wave32 shuffle reductions over 40 classes
__global__ __launch_bounds__(256) void logsoftmax_kernel(
    const float* __restrict__ logits, float* __restrict__ out, int N)
{
    int node = blockIdx.x * (blockDim.x >> 5) + (threadIdx.x >> 5);
    int lane = threadIdx.x & 31;
    if (node >= N) return;
    const float* row = logits + (long)node * NCLASS;
    float x1 = (lane < NCLASS) ? row[lane] : -3.4e38f;
    float x2 = (lane + 32 < NCLASS) ? row[lane + 32] : -3.4e38f;
    float m = fmaxf(x1, x2);
    for (int off = 16; off; off >>= 1) m = fmaxf(m, __shfl_xor(m, off, 32));
    float s = 0.f;
    if (lane < NCLASS)      s += __expf(x1 - m);
    if (lane + 32 < NCLASS) s += __expf(x2 - m);
    for (int off = 16; off; off >>= 1) s += __shfl_xor(s, off, 32);
    float ls = __logf(s);
    if (lane < NCLASS)      out[(long)node * NCLASS + lane]      = x1 - m - ls;
    if (lane + 32 < NCLASS) out[(long)node * NCLASS + lane + 32] = x2 - m - ls;
}

// out2 = h copy; out3 = concat([h, logits], axis=1)
__global__ void outputs_kernel(const float* __restrict__ h, const float* __restrict__ logits,
                               float* __restrict__ out_h, float* __restrict__ out_cat, long N)
{
    long idx = (long)blockIdx.x * blockDim.x + threadIdx.x;
    long total = N * (NHID + NCLASS);
    if (idx >= total) return;
    long n = idx / (NHID + NCLASS);
    int f = (int)(idx % (NHID + NCLASS));
    float v = (f < NHID) ? h[n * NHID + f] : logits[n * NCLASS + (f - NHID)];
    out_cat[idx] = v;
    if (f < NHID) out_h[n * NHID + f] = v;
}

// ---------------------------------------------------------------------------
extern "C" void kernel_launch(void* const* d_in, const int* in_sizes, int n_in,
                              void* d_out, int out_size, void* d_ws, size_t ws_size,
                              hipStream_t stream)
{
    const float* x     = (const float*)d_in[0];
    const int*   rows  = (const int*)d_in[1];
    const int*   cols  = (const int*)d_in[2];
    const float* vals  = (const float*)d_in[3];
    const float* fc1_w = (const float*)d_in[4];
    const float* fc1_b = (const float*)d_in[5];
    const float* convw = (const float*)d_in[6];
    const float* fc2_w = (const float*)d_in[7];
    const float* fc2_b = (const float*)d_in[8];
    const int nE = in_sizes[1];

    // ---- workspace carve-out ----
    char* ws = (char*)d_ws;
    size_t off = 0;
    auto carve = [&](size_t bytes) -> char* {
        char* p = ws + off;
        off += (bytes + 255) & ~(size_t)255;
        return p;
    };
    const long HN = (long)NNODES * NHID;
    float*  h0     = (float*)carve(HN * 4);
    float*  hcur   = (float*)carve(HN * 4);
    float*  hi     = (float*)carve(HN * 4);
    float*  logits = (float*)carve((long)NNODES * NCLASS * 4);
    __bf16* pfc1   = (__bf16*)carve((size_t)(NFEAT / 32) * 8 * 512 * 2);
    __bf16* pconv[NLAYER];
    for (int i = 0; i < NLAYER; ++i) pconv[i] = (__bf16*)carve((size_t)(NHID / 32) * 8 * 512 * 2);
    __bf16* pfc2   = (__bf16*)carve((size_t)(NHID / 32) * 3 * 512 * 2);

    // ---- pack weights to bf16 WMMA B-fragments ----
    {
        int tot = (NFEAT / 32) * 8 * 512;
        pack_w_kernel<<<(tot + 255) / 256, 256, 0, stream>>>(fc1_w, pfc1, NFEAT, NHID, 8);
        tot = (NHID / 32) * 8 * 512;
        for (int i = 0; i < NLAYER; ++i)
            pack_w_kernel<<<(tot + 255) / 256, 256, 0, stream>>>(
                convw + (long)i * NHID * NHID, pconv[i], NHID, NHID, 8);
        tot = (NHID / 32) * 3 * 512;
        pack_w_kernel<<<(tot + 255) / 256, 256, 0, stream>>>(fc2_w, pfc2, NHID, NCLASS, 3);
    }

    const int strips = NNODES / 16;                 // 6250 (exact)
    const int gemmBlocks = (strips + 7) / 8;
    const long hnElems = HN;
    const int ewBlocks = (int)((hnElems + 255) / 256);

    // ---- fc1: h0 = relu(x @ fc1_w + b) ----  (LDS: 16*8*1024 = 128KB)
    gemm_wmma_kernel<8, 0, 16><<<gemmBlocks, 256, 16 * 8 * 1024, stream>>>(
        x, NFEAT, pfc1, h0, NHID, fc1_b, nullptr, 0.0f, NNODES, NHID);

    // ---- GCNII layers ----
    const int spmmWaves  = (nE + SPMM_SPAN - 1) / SPMM_SPAN;
    const int spmmBlocks = (spmmWaves + 7) / 8;
    for (int i = 0; i < NLAYER; ++i) {
        float theta = logf(LAMDA_ / (float)(i + 1) + 1.0f);
        const float* hin = (i == 0) ? h0 : hcur;
        zero_kernel<<<ewBlocks, 256, 0, stream>>>(hi, hnElems);
        spmm_kernel<<<spmmBlocks, 256, 0, stream>>>(rows, cols, vals, hin, hi, nE);
        support_kernel<<<ewBlocks, 256, 0, stream>>>(hi, h0, hnElems);
        gemm_wmma_kernel<8, 1, 4><<<gemmBlocks, 256, 4 * 8 * 1024, stream>>>(
            hi, NHID, pconv[i], hcur, NHID, nullptr, hi, theta, NNODES, NHID);
    }

    // ---- fc2: logits = h @ fc2_w + b ----  (LDS: 4*3*1024 = 12KB)
    gemm_wmma_kernel<3, 2, 4><<<gemmBlocks, 256, 4 * 3 * 1024, stream>>>(
        hcur, NHID, pfc2, logits, NHID, fc2_b, nullptr, 0.0f, NNODES, NCLASS);

    // ---- outputs: [log_softmax, h, concat(h, logits)] ----
    float* out_ls  = (float*)d_out;
    float* out_h   = out_ls + (long)NNODES * NCLASS;
    float* out_cat = out_h + HN;
    logsoftmax_kernel<<<(NNODES + 7) / 8, 256, 0, stream>>>(logits, out_ls, NNODES);
    long catTot = (long)NNODES * (NHID + NCLASS);
    outputs_kernel<<<(int)((catTot + 255) / 256), 256, 0, stream>>>(hcur, logits, out_h, out_cat, NNODES);
}
